// FiconnModel_26723286516363
// MI455X (gfx1250) — matrix operand order repaired and checked by reading, not compiled, not run on hardware
//
#include <hip/hip_runtime.h>
#include <math.h>
#include <stdint.h>

#define NDIM 128
#define MTILE 32          // batch rows per workgroup
#define KST  136          // packed bf16 plane row stride (halves): 128 data + 8 pad
#define ZPL  (MTILE * KST)   // z plane size (halves) = 4352
#define UPL  (NDIM * KST)    // U plane size (halves) = 17408
#define UB   132          // build-kernel f32 LDS row stride

typedef __attribute__((ext_vector_type(8)))  float v8f;
typedef __attribute__((ext_vector_type(4)))  unsigned int v4u;
typedef __attribute__((ext_vector_type(8)))  unsigned int v8u;
typedef __attribute__((ext_vector_type(8)))  int v8i;
typedef __attribute__((ext_vector_type(4)))  int v4i;
typedef __attribute__((ext_vector_type(16))) __bf16 v16bf;

// ---------------- bf16 helpers (manual RNE, exact hi+lo split) ----------------
static __device__ __forceinline__ unsigned short f2bf(float f) {
  unsigned u = __builtin_bit_cast(unsigned, f);
  unsigned r = u + 0x7FFFu + ((u >> 16) & 1u);  // round to nearest even
  return (unsigned short)(r >> 16);
}
static __device__ __forceinline__ float bf2f(unsigned short h) {
  return __builtin_bit_cast(float, (unsigned)h << 16);
}

static __device__ __forceinline__ v8f wmma_bf16(v16bf a, v16bf b, v8f c) {
  // D(16x16 f32) = A(16x32 bf16) * B(32x16 bf16) + C  — XDL fast path
  return __builtin_amdgcn_wmma_f32_16x16x32_bf16(false, a, false, b, (short)0, c,
                                                 false, false);
}

static __device__ __forceinline__ v16bf negbf(v16bf a) {  // exact bf16 negation
  v8u u = __builtin_bit_cast(v8u, a);
  u ^= 0x80008000u;
  return __builtin_bit_cast(v16bf, u);
}

// A-fragment (16x32, M x K): lane(m=lane&15, h=lane>>4); VGPR j<4: K=8h+2j(+1),
// VGPR j>=4: K=16+8h+2(j-4)(+1). Plane layout [m][k] bf16, k fastest -> 2x b128.
static __device__ __forceinline__ v16bf load_afrag(const unsigned short* plane,
                                                   int m, int k0, int h) {
  const unsigned short* p = plane + m * KST + k0 + 8 * h;
  v4u lo = *(const v4u*)p;         // K = k0+8h .. +7
  v4u hi = *(const v4u*)(p + 16);  // K = k0+16+8h .. +7
  v8u u = {lo.x, lo.y, lo.z, lo.w, hi.x, hi.y, hi.z, hi.w};
  return __builtin_bit_cast(v16bf, u);
}

// B-fragment (32x16, K x N): lane(n=lane&15, h); VGPR j<4: K=8h+2j(+1),
// VGPR j>=4: K=16+8h+2(j-4)(+1).
// Plane layout [n][k] bf16, k fastest -> 2x b128 at byte offsets +0 and +32.
static __device__ __forceinline__ v16bf load_bfrag(const unsigned short* plane,
                                                   int n, int k0, int h) {
  const unsigned short* p = plane + n * KST + k0 + 8 * h;
  v4u lo = *(const v4u*)p;
  v4u hi = *(const v4u*)(p + 16);
  v8u u = {lo.x, lo.y, lo.z, lo.w, hi.x, hi.y, hi.z, hi.w};
  return __builtin_bit_cast(v16bf, u);
}

// ---------------------------------------------------------------------------
// Tensor Data Mover: stage one 128x128 plane (row-major, data_size bytes/elem)
// into LDS with +4 dwords pad per row (-> 136-half rows for bf16 planes).
// D# per CDNA5 ISA; tracked with TENSORcnt.
// ---------------------------------------------------------------------------
static __device__ __forceinline__ void tdm_load_plane_128x128_bf16(
    const unsigned short* gsrc, unsigned lds_byte_off) {
  unsigned long long ga = (unsigned long long)(uintptr_t)gsrc;
  v4u g0;
  g0.x = 1u;                                   // count=1, is_restore=0, no gather
  g0.y = lds_byte_off;                         // lds_addr [63:32]
  g0.z = (unsigned)(ga & 0xFFFFFFFFu);         // global_addr [95:64]
  g0.w = (unsigned)((ga >> 32) & 0x01FFFFFFu)  // global_addr [120:96]
         | (2u << 30);                         // type=2 ("image")
  v8i g1;
  g1[0] = (1 << 16)     // data_size = 2 bytes (bf16)
        | (1 << 20)     // pad_enable
        | (5 << 22)     // pad_interval: 64 dwords (= 128 halves)
        | (3 << 25);    // pad_amount: 4 dwords (= 8 halves) -> row stride 136 halves
  g1[1] = (128 << 16);  // tensor_dim0 = 128 elems (low 16)
  g1[2] = (128 << 16);  // tensor_dim0 hi=0; tensor_dim1 = 128 (low 16)
  g1[3] = (128 << 16);  // tensor_dim1 hi=0; tile_dim0 = 128
  g1[4] = 128;          // tile_dim1 = 128; tile_dim2 = 0
  g1[5] = 128;          // tensor_dim0_stride = 128 (low 32)
  g1[6] = (128 << 16);  // stride hi=0; tensor_dim1_stride = 128 (low 16)
  g1[7] = 0;
  v4i gz4 = {0, 0, 0, 0};
  v8i gz8 = {0, 0, 0, 0, 0, 0, 0, 0};
  __builtin_amdgcn_tensor_load_to_lds(g0, g1, gz4, gz4, gz8, 0);
}

// ---------------------------------------------------------------------------
// Kernel 1: blocks 0..2 build the Clements-mesh unitaries (f32 in LDS), then
// emit transposed bf16 hi/lo planes U^T[n][k] to ws; block 3 emits W^T planes.
// ws (halves): stage s -> planes [4][16384] at s*4*16384; W planes at 12*16384.
// ---------------------------------------------------------------------------
__global__ __launch_bounds__(1024, 1) void build_unitary_kernel(
    const float* __restrict__ mzi1, const float* __restrict__ op1,
    const float* __restrict__ mzi2, const float* __restrict__ op2,
    const float* __restrict__ mzi3, const float* __restrict__ op3,
    const float* __restrict__ W, unsigned short* __restrict__ ws) {
  const int tid = threadIdx.x;

  if (blockIdx.x == 3) {  // W^T planes: [j][k] = W[j][k] (re) / W[j][128+k] (im)
    unsigned short* wp = ws + 12u * NDIM * NDIM;
    for (int i = tid; i < NDIM * NDIM; i += blockDim.x) {
      int j = i >> 7, k = i & 127;
      float re = W[j * 256 + k];
      float im = W[j * 256 + 128 + k];
      unsigned short rh = f2bf(re);
      unsigned short ih = f2bf(im);
      wp[0 * NDIM * NDIM + i] = rh;
      wp[1 * NDIM * NDIM + i] = f2bf(re - bf2f(rh));
      wp[2 * NDIM * NDIM + i] = ih;
      wp[3 * NDIM * NDIM + i] = f2bf(im - bf2f(ih));
    }
    return;
  }

  extern __shared__ float lds[];
  float* ure = lds;                 // 128*UB
  float* uim = ure + NDIM * UB;     // 128*UB
  float* ct  = uim + NDIM * UB;     // 64
  float* st  = ct + 64;
  float* er  = st + 64;
  float* ei  = er + 64;

  const float* mzi = (blockIdx.x == 0) ? mzi1 : ((blockIdx.x == 1) ? mzi2 : mzi3);
  const float* op  = (blockIdx.x == 0) ? op1  : ((blockIdx.x == 1) ? op2  : op3);
  unsigned short* gp = ws + (size_t)blockIdx.x * 4u * NDIM * NDIM;

  // U = I
  for (int i = tid; i < NDIM * NDIM; i += blockDim.x) {
    int r = i >> 7, c = i & 127;
    ure[r * UB + c] = (r == c) ? 1.f : 0.f;
    uim[r * UB + c] = 0.f;
  }
  __syncthreads();

  for (int layer = 0; layer < NDIM; ++layer) {
    const int par = layer & 1;
    const int m = par ? 63 : 64;
    const int idx = 127 * layer + par;  // phase offset (even layer: 128, odd: 126)
    if (tid < m) {
      float th = mzi[idx + 2 * tid];
      float ph = mzi[idx + 2 * tid + 1];
      ct[tid] = __cosf(th);
      st[tid] = __sinf(th);
      er[tid] = __cosf(ph);
      ei[tid] = __sinf(ph);
    }
    __syncthreads();
    const int items = NDIM * m;
    for (int i = tid; i < items; i += blockDim.x) {
      int row = i / m;
      int p = i - row * m;
      int o0 = row * UB + par + 2 * p;
      float u0r = ure[o0],     u0i = uim[o0];
      float u1r = ure[o0 + 1], u1i = uim[o0 + 1];
      float cc = ct[p], ss = st[p];
      float ar = cc * er[p], ai = cc * ei[p];  // ct*e^{i phi}
      float n0r = u0r * ar - u0i * ai + ss * u1r;
      float n0i = u0r * ai + u0i * ar + ss * u1i;
      float n1r = ss * u0r - (ar * u1r + ai * u1i);
      float n1i = ss * u0i - (ar * u1i - ai * u1r);
      ure[o0] = n0r;     uim[o0] = n0i;
      ure[o0 + 1] = n1r; uim[o0 + 1] = n1i;
    }
    __syncthreads();
  }

  // fold output phases; write transposed hi/lo planes: plane[n*128+k] = U[k][n]
  for (int i = tid; i < NDIM * NDIM; i += blockDim.x) {
    int n = i >> 7, k = i & 127;
    float pc = __cosf(op[n]), ps = __sinf(op[n]);
    float re0 = ure[k * UB + n], im0 = uim[k * UB + n];
    float re = re0 * pc - im0 * ps;
    float im = re0 * ps + im0 * pc;
    unsigned short rh = f2bf(re);
    unsigned short ih = f2bf(im);
    gp[0 * NDIM * NDIM + i] = rh;
    gp[1 * NDIM * NDIM + i] = f2bf(re - bf2f(rh));
    gp[2 * NDIM * NDIM + i] = ih;
    gp[3 * NDIM * NDIM + i] = f2bf(im - bf2f(ih));
  }
}

// ---------------------------------------------------------------------------
// Kernel 2: fused pipeline per 32-row batch tile on v_wmma_f32_16x16x32_bf16
// with exact hi/lo split (3 WMMAs per real product ~= fp32 accuracy).
// z buffers + U planes are packed-bf16 LDS planes [reH, reL, imH, imL].
// ---------------------------------------------------------------------------

static __device__ __forceinline__ void split_store(unsigned short* zo, int off,
                                                   float re, float im) {
  unsigned short rh = f2bf(re);
  unsigned short ih = f2bf(im);
  zo[0 * ZPL + off] = rh;
  zo[1 * ZPL + off] = f2bf(re - bf2f(rh));
  zo[2 * ZPL + off] = ih;
  zo[3 * ZPL + off] = f2bf(im - bf2f(ih));
}

// zout = (optionally nofu of) zin @ (Ure + i Uim)
static __device__ __forceinline__ void complex_stage(
    const unsigned short* __restrict__ zin, const unsigned short* __restrict__ U,
    unsigned short* __restrict__ zout,
    const float* __restrict__ atab, const float* __restrict__ btab, int tid) {
  const int wave = tid >> 5, lane = tid & 31;
  const int ml = lane & 15, h = lane >> 4;
  const int rowbase = (wave >> 2) * 16;  // 2 row-subtiles
  const int ntbase = (wave & 3) * 2;     // 2 N-tiles per wave

  v8f accR[2] = {};
  v8f accI[2] = {};

  for (int k0 = 0; k0 < NDIM; k0 += 32) {
    const int m = rowbase + ml;
    v16bf arH = load_afrag(zin + 0 * ZPL, m, k0, h);
    v16bf arL = load_afrag(zin + 1 * ZPL, m, k0, h);
    v16bf aiH = load_afrag(zin + 2 * ZPL, m, k0, h);
    v16bf aiL = load_afrag(zin + 3 * ZPL, m, k0, h);
    v16bf naH = negbf(aiH);
    v16bf naL = negbf(aiL);
#pragma unroll
    for (int t = 0; t < 2; ++t) {
      const int nc = (ntbase + t) * 16 + ml;
      v16bf brH = load_bfrag(U + 0 * UPL, nc, k0, h);
      v16bf brL = load_bfrag(U + 1 * UPL, nc, k0, h);
      v16bf biH = load_bfrag(U + 2 * UPL, nc, k0, h);
      v16bf biL = load_bfrag(U + 3 * UPL, nc, k0, h);
      // Re += Ar*Br - Ai*Bi  (hi*hi + hi*lo + lo*hi per product)
      accR[t] = wmma_bf16(arH, brH, accR[t]);
      accR[t] = wmma_bf16(arH, brL, accR[t]);
      accR[t] = wmma_bf16(arL, brH, accR[t]);
      accR[t] = wmma_bf16(naH, biH, accR[t]);
      accR[t] = wmma_bf16(naH, biL, accR[t]);
      accR[t] = wmma_bf16(naL, biH, accR[t]);
      // Im += Ar*Bi + Ai*Br
      accI[t] = wmma_bf16(arH, biH, accI[t]);
      accI[t] = wmma_bf16(arH, biL, accI[t]);
      accI[t] = wmma_bf16(arL, biH, accI[t]);
      accI[t] = wmma_bf16(aiH, brH, accI[t]);
      accI[t] = wmma_bf16(aiH, brL, accI[t]);
      accI[t] = wmma_bf16(aiL, brH, accI[t]);
    }
  }

#pragma unroll
  for (int t = 0; t < 2; ++t) {
    const int nc = (ntbase + t) * 16 + ml;
    float av = 0.f, bw = 0.f;
    if (atab) { av = atab[nc]; bw = btab[nc]; }
#pragma unroll
    for (int v = 0; v < 8; ++v) {
      const int row = rowbase + v + 8 * h;
      float re = accR[t][v], im = accI[t][v];
      if (atab) {  // nofu in f32 on accumulators; sqrt(exp(x)) == exp(x/2)
        float inten = re * re + im * im + 1e-8f;
        float inv = __builtin_amdgcn_rcpf(1.f + bw * inten);  // v_rcp_f32
        float s = __expf(-0.5f * av * inv);                   // v_exp_f32
        re *= s; im *= s;
      }
      split_store(zout, row * KST + nc, re, im);
    }
  }
}

// out = Re(z)@Wre^T + Im(z)@Wim^T + b   (f32 result to global)
static __device__ __forceinline__ void final_stage(
    const unsigned short* __restrict__ zin, const unsigned short* __restrict__ Wp,
    const float* __restrict__ bv, float* __restrict__ out, size_t rowbase_g,
    int tid) {
  const int wave = tid >> 5, lane = tid & 31;
  const int ml = lane & 15, h = lane >> 4;
  const int rowbase = (wave >> 2) * 16;
  const int ntbase = (wave & 3) * 2;

  v8f acc[2] = {};
  for (int k0 = 0; k0 < NDIM; k0 += 32) {
    const int m = rowbase + ml;
    v16bf arH = load_afrag(zin + 0 * ZPL, m, k0, h);
    v16bf arL = load_afrag(zin + 1 * ZPL, m, k0, h);
    v16bf aiH = load_afrag(zin + 2 * ZPL, m, k0, h);
    v16bf aiL = load_afrag(zin + 3 * ZPL, m, k0, h);
#pragma unroll
    for (int t = 0; t < 2; ++t) {
      const int nc = (ntbase + t) * 16 + ml;
      v16bf brH = load_bfrag(Wp + 0 * UPL, nc, k0, h);
      v16bf brL = load_bfrag(Wp + 1 * UPL, nc, k0, h);
      v16bf biH = load_bfrag(Wp + 2 * UPL, nc, k0, h);
      v16bf biL = load_bfrag(Wp + 3 * UPL, nc, k0, h);
      acc[t] = wmma_bf16(arH, brH, acc[t]);
      acc[t] = wmma_bf16(arH, brL, acc[t]);
      acc[t] = wmma_bf16(arL, brH, acc[t]);
      acc[t] = wmma_bf16(aiH, biH, acc[t]);
      acc[t] = wmma_bf16(aiH, biL, acc[t]);
      acc[t] = wmma_bf16(aiL, biH, acc[t]);
    }
  }
#pragma unroll
  for (int t = 0; t < 2; ++t) {
    const int nc = (ntbase + t) * 16 + ml;
    const float bb = bv[nc];
#pragma unroll
    for (int v = 0; v < 8; ++v) {
      const int row = rowbase + v + 8 * h;
      out[(rowbase_g + row) * NDIM + nc] = acc[t][v] + bb;
    }
  }
}

// LDS layout (halves): zA[4*ZPL] | zB[4*ZPL] | U[4*UPL] | tables (f32)
#define LDSOFF_ZA 0u
#define LDSOFF_ZB (4u * ZPL)
#define LDSOFF_U  (8u * ZPL)
#define LDS_HALVES (8u * ZPL + 4u * UPL)

__global__ __launch_bounds__(256, 1) void ficonn_fused_kernel(
    const float* __restrict__ x, const float* __restrict__ input_phases,
    const float* __restrict__ alpha1, const float* __restrict__ beta1,
    const float* __restrict__ alpha2, const float* __restrict__ beta2,
    const float* __restrict__ bias,
    const unsigned short* __restrict__ ws, float* __restrict__ out) {
  extern __shared__ unsigned short ldsu[];
  unsigned short* zA = ldsu + LDSOFF_ZA;
  unsigned short* zB = ldsu + LDSOFF_ZB;
  unsigned short* Up = ldsu + LDSOFF_U;
  float* tab = (float*)(ldsu + LDS_HALVES);  // 7*128 f32
  float* ipc = tab;        float* ips = tab + 128;
  float* a1t = tab + 256;  float* b1t = tab + 384;
  float* a2t = tab + 512;  float* b2t = tab + 640;
  float* bvt = tab + 768;

  const int tid = threadIdx.x;
  const size_t rowbase_g = (size_t)blockIdx.x * MTILE;
  const bool issuer = (tid < 32);  // wave 0 drives the TDM
  const unsigned NN = NDIM * NDIM;

  // Kick off TDM staging of the 4 U1 planes while tables / x load proceed.
  if (issuer) {
#pragma unroll
    for (int p = 0; p < 4; ++p)
      tdm_load_plane_128x128_bf16(ws + (0u * 4u + p) * NN,
                                  (LDSOFF_U + p * UPL) * 2u);
  }
  if (tid < NDIM) {
    float ip = input_phases[tid];
    ipc[tid] = __cosf(ip);
    ips[tid] = __sinf(ip);
    a1t[tid] = fminf(fmaxf(alpha1[tid], 0.f), 10.f);
    b1t[tid] = fminf(fmaxf(beta1[tid], 0.f), 10.f);
    a2t[tid] = fminf(fmaxf(alpha2[tid], 0.f), 10.f);
    b2t[tid] = fminf(fmaxf(beta2[tid], 0.f), 10.f);
    bvt[tid] = bias[tid];
  }
  __syncthreads();

  // x tile -> zA planes with input phases applied
  for (int i = tid; i < MTILE * NDIM; i += 256) {
    int r = i >> 7, c = i & 127;
    float xv = x[(rowbase_g + r) * NDIM + c];
    split_store(zA, r * KST + c, xv * ipc[c], xv * ips[c]);
  }
  if (issuer) __builtin_amdgcn_s_wait_tensorcnt(0);
  __syncthreads();

  // Stage 1: zB = nofu(zA @ U1)
  complex_stage(zA, Up, zB, a1t, b1t, tid);
  __syncthreads();

  if (issuer) {
#pragma unroll
    for (int p = 0; p < 4; ++p)
      tdm_load_plane_128x128_bf16(ws + (1u * 4u + p) * NN,
                                  (LDSOFF_U + p * UPL) * 2u);
    __builtin_amdgcn_s_wait_tensorcnt(0);
  }
  __syncthreads();

  // Stage 2: zA = nofu(zB @ U2)
  complex_stage(zB, Up, zA, a2t, b2t, tid);
  __syncthreads();

  if (issuer) {
#pragma unroll
    for (int p = 0; p < 4; ++p)
      tdm_load_plane_128x128_bf16(ws + (2u * 4u + p) * NN,
                                  (LDSOFF_U + p * UPL) * 2u);
    __builtin_amdgcn_s_wait_tensorcnt(0);
  }
  __syncthreads();

  // Stage 3: zB = zA @ U3 (no nofu)
  complex_stage(zA, Up, zB, nullptr, nullptr, tid);
  __syncthreads();

  if (issuer) {
#pragma unroll
    for (int p = 0; p < 4; ++p)
      tdm_load_plane_128x128_bf16(ws + (3u * 4u + p) * NN,  // W^T planes
                                  (LDSOFF_U + p * UPL) * 2u);
    __builtin_amdgcn_s_wait_tensorcnt(0);
  }
  __syncthreads();

  // Final: out = Re(zB)@Wre^T + Im(zB)@Wim^T + b
  final_stage(zB, Up, bvt, out, rowbase_g, tid);
}

extern "C" void kernel_launch(void* const* d_in, const int* in_sizes, int n_in,
                              void* d_out, int out_size, void* d_ws, size_t ws_size,
                              hipStream_t stream) {
  const float* x    = (const float*)d_in[0];
  const float* ip   = (const float*)d_in[1];
  const float* mzi1 = (const float*)d_in[2];
  const float* op1  = (const float*)d_in[3];
  const float* a1   = (const float*)d_in[4];
  const float* b1   = (const float*)d_in[5];
  const float* mzi2 = (const float*)d_in[6];
  const float* op2  = (const float*)d_in[7];
  const float* a2   = (const float*)d_in[8];
  const float* b2   = (const float*)d_in[9];
  const float* mzi3 = (const float*)d_in[10];
  const float* op3  = (const float*)d_in[11];
  const float* W    = (const float*)d_in[12];
  const float* bias = (const float*)d_in[13];
  float* out = (float*)d_out;
  unsigned short* ws = (unsigned short*)d_ws;  // 16 planes x 16384 halves = 512 KB

  const size_t lds_build = (size_t)(2 * NDIM * UB + 4 * 64) * sizeof(float);
  build_unitary_kernel<<<4, 1024, lds_build, stream>>>(mzi1, op1, mzi2, op2, mzi3,
                                                       op3, W, ws);

  const int B = in_sizes[0] / NDIM;  // 65536
  const size_t lds_gemm = (size_t)LDS_HALVES * 2u + 7u * 128u * sizeof(float);
  ficonn_fused_kernel<<<B / MTILE, 256, lds_gemm, stream>>>(
      x, ip, a1, b1, a2, b2, bias, ws, out);
}
